// FadeFeatureAlignment_18794776887360
// MI455X (gfx1250) — compile-verified, matrix operand order checked by reading, not run on hardware
//
#include <hip/hip_runtime.h>
#include <hip/hip_bf16.h>

// ---------------------------------------------------------------------------
// FADE/CARAFE feature alignment for MI455X (gfx1250, wave32, WMMA).
// Projection GEMMs (8.6 GFLOP each) run on v_wmma_f32_16x16x32_bf16 with the
// whole B pack staged in LDS (one coalesced copy per workgroup, then
// ds_load_b128 fragment reads -> short waits before each WMMA instead of
// full global-load latency).
// ---------------------------------------------------------------------------

typedef __bf16 bf16;
typedef __attribute__((ext_vector_type(16))) __bf16 v16bf;
typedef __attribute__((ext_vector_type(8)))  __bf16 v8bf;
typedef __attribute__((ext_vector_type(8)))  float  v8f;
typedef __attribute__((ext_vector_type(4)))  float  v4f;

union FragBF { v16bf v; v8bf h[2]; };

// K index inside a 16x32 A-fragment (ISA 7.12.2, 16-bit A layout):
// lane half h: VGPR0..3 hold K = 8h..8h+7, VGPR4..7 hold K = 16+8h..16+8h+7.
__device__ __forceinline__ int kmap(int h, int e) {
  return (e < 8) ? (8 * h + e) : (16 + 8 * h + (e - 8));
}

// --------------------------- weight packing --------------------------------
// Pack row-major W[K][N] (f32) into bf16 B-fragments so each lane's v16bf is
// one contiguous 32-byte chunk: pack[((nt*KT + kt)*32 + lane)*16 + e].
__global__ void pack_weights_kernel(const float* __restrict__ w,
                                    bf16* __restrict__ out, int K, int N) {
  int idx = blockIdx.x * blockDim.x + threadIdx.x;
  if (idx >= K * N) return;
  int e    = idx & 15;
  int lane = (idx >> 4) & 31;
  int t    = idx >> 9;          // nt*KT + kt
  int KT   = K >> 5;
  int kt   = t % KT;
  int nt   = t / KT;
  int n = nt * 16 + (lane & 15);
  int k = kt * 32 + kmap(lane >> 4, e);
  out[idx] = (bf16)w[(size_t)k * N + n];
}

// --------------------------- fragment loaders ------------------------------
__device__ __forceinline__ v16bf load_a_bf16(const bf16* A, int lda, int row0,
                                             int k0, int lane) {
  int r = lane & 15, h = lane >> 4;
  const bf16* p = A + (size_t)(row0 + r) * lda + k0 + 8 * h;
  FragBF f;
  f.h[0] = *(const v8bf*)p;
  f.h[1] = *(const v8bf*)(p + 16);
  return f.v;
}

__device__ __forceinline__ v16bf load_a_f32(const float* A, int lda, int row0,
                                            int k0, int lane) {
  int r = lane & 15, h = lane >> 4;
  const float* p0 = A + (size_t)(row0 + r) * lda + k0 + 8 * h;
  const float* p1 = p0 + 16;
  v16bf out;
#pragma unroll
  for (int e = 0; e < 8; ++e) out[e] = (bf16)p0[e];
#pragma unroll
  for (int e = 0; e < 8; ++e) out[8 + e] = (bf16)p1[e];
  return out;
}

// B fragment t = nt*KT+kt from LDS-staged pack: 1 KB per fragment,
// 32 B per lane -> two ds_load_b128.
__device__ __forceinline__ v16bf load_b_lds(const bf16* lds, int t, int lane) {
  const v8bf* p = (const v8bf*)lds + (size_t)t * 64 + lane * 2;
  FragBF f;
  f.h[0] = p[0];
  f.h[1] = p[1];
  return f.v;
}

// Coalesced global -> LDS staging of the B pack (16-byte chunks).
__device__ __forceinline__ void stage_b(const bf16* __restrict__ g,
                                        bf16* lds, int bytes) {
  int chunks = bytes >> 4;
  for (int c = threadIdx.x; c < chunks; c += blockDim.x)
    ((v8bf*)lds)[c] = ((const v8bf*)g)[c];
  __syncthreads();
}

// --------------------------- gate ------------------------------------------
__global__ void gate_kernel(const float* __restrict__ coarse,
                            const float* __restrict__ gw,
                            const float* __restrict__ gb,
                            float* __restrict__ gate, int npix, int C) {
  int p = blockIdx.x * blockDim.x + threadIdx.x;
  if (p >= npix) return;
  const float* x = coarse + (size_t)p * C;
  float acc = 0.f;
  for (int c = 0; c < C; c += 4) {
    v4f xv = *(const v4f*)(x + c);
    v4f wv = *(const v4f*)(gw + c);
    acc += xv.x * wv.x + xv.y * wv.y + xv.z * wv.z + xv.w * wv.w;
  }
  acc += gb[0];
  gate[p] = 1.f / (1.f + __expf(-acc));
}

// --------------------------- embedding GEMMs (A = f32) ---------------------
// C[M,N] = A[M,K] x Bpack + (bias); one wave per 16-row tile; whole B pack
// staged in LDS (16-32 KB).
template <int KT, int NT, bool BIAS>
__global__ __launch_bounds__(256, 1)
void gemm_a32_kernel(const float* __restrict__ A,
                     const bf16* __restrict__ Bp,
                     const float* __restrict__ bias,
                     float* __restrict__ C,
                     int Mtiles, int lda, int N) {
  __shared__ bf16 bsh[NT * KT * 512];
  stage_b(Bp, bsh, NT * KT * 1024);

  int wid  = threadIdx.x >> 5;
  int lane = threadIdx.x & 31;
  int wave = blockIdx.x * 8 + wid;
  if (wave >= Mtiles) return;               // wave-uniform: EXEC stays all-1s
  int row0 = wave * 16;
  v8f acc[NT];
#pragma unroll
  for (int n = 0; n < NT; ++n) { v8f z = {}; acc[n] = z; }
  for (int kt = 0; kt < KT; ++kt) {
    v16bf a = load_a_f32(A, lda, row0, kt * 32, lane);
#pragma unroll
    for (int n = 0; n < NT; ++n) {
      v16bf b = load_b_lds(bsh, n * KT + kt, lane);
      acc[n] = __builtin_amdgcn_wmma_f32_16x16x32_bf16(
          false, a, false, b, (short)0, acc[n], false, false);
    }
  }
  int r = lane & 15, h = lane >> 4;
#pragma unroll
  for (int n = 0; n < NT; ++n) {
    int col = n * 16 + r;
    float bv = BIAS ? bias[col] : 0.f;
#pragma unroll
    for (int g = 0; g < 8; ++g) {
      int row = row0 + g + 8 * h;           // C layout: lanes0-15 M=g, 16-31 M=g+8
      C[(size_t)row * N + col] = acc[n][g] + bv;
    }
  }
}

// --------------------------- shared 3x3 content conv -----------------------
// src [Bn,Hh,Ww,64] -> out [Bn,Hh,Ww,25]; per-tap 64x25 weight slice in LDS.
__global__ void conv3x3_kernel(const float* __restrict__ src,
                               const float* __restrict__ w,     // [3,3,64,25]
                               const float* __restrict__ bias,  // [25]
                               float* __restrict__ out, int Bn, int Hh, int Ww) {
  __shared__ float wl[64 * 25];
  int tid = threadIdx.x;
  int p = blockIdx.x * blockDim.x + tid;
  int npix = Bn * Hh * Ww;
  bool valid = p < npix;
  int b = 0, i = 0, j = 0;
  if (valid) {
    b = p / (Hh * Ww);
    int rem = p % (Hh * Ww);
    i = rem / Ww;
    j = rem % Ww;
  }
  float acc[25];
#pragma unroll
  for (int k = 0; k < 25; ++k) acc[k] = bias[k];
  for (int t = 0; t < 9; ++t) {
    __syncthreads();
    for (int idx = tid; idx < 64 * 25; idx += blockDim.x)
      wl[idx] = w[t * 64 * 25 + idx];
    __syncthreads();
    if (valid) {
      int y = i + t / 3 - 1, x = j + t % 3 - 1;
      if (y >= 0 && y < Hh && x >= 0 && x < Ww) {
        const float* s = src + (((size_t)b * Hh + y) * Ww + x) * 64;
        for (int c = 0; c < 64; ++c) {
          float xv = s[c];
#pragma unroll
          for (int k = 0; k < 25; ++k) acc[k] += xv * wl[c * 25 + k];
        }
      }
    }
  }
  if (valid) {
    float* o = out + (size_t)p * 25;
#pragma unroll
    for (int k = 0; k < 25; ++k) o[k] = acc[k];
  }
}

// --------------------------- CARAFE + fused softmax ------------------------
// One thread per fine pixel; coarse (16.8 MB) is L2-resident on 192 MB L2.
// Writes cu row-major bf16 == A-fragment source format for GEMM1.
__global__ void carafe_kernel(const float* __restrict__ coarse,   // [4,64,64,256]
                              const float* __restrict__ flog,     // [65536,25]
                              const float* __restrict__ clog,     // [16384,25]
                              bf16* __restrict__ cu) {            // [65536,256]
  int p = blockIdx.x * blockDim.x + threadIdx.x;
  if (p >= 4 * 128 * 128) return;
  int b = p >> 14, rem = p & 16383, i = rem >> 7, j = rem & 127;
  int hc = i >> 1, wc = j >> 1;
  const float* fl = flog + (size_t)p * 25;
  const float* cl = clog + (size_t)(((b << 6) + hc) * 64 + wc) * 25;
  float m[25];
  float mx = -3.0e38f;
#pragma unroll
  for (int k = 0; k < 25; ++k) { m[k] = fl[k] + cl[k]; mx = fmaxf(mx, m[k]); }
  float s = 0.f;
#pragma unroll
  for (int k = 0; k < 25; ++k) { m[k] = __expf(m[k] - mx); s += m[k]; }
  float inv = 1.f / s;
#pragma unroll
  for (int k = 0; k < 25; ++k) m[k] *= inv;

  int offk[25];
  bool vk[25];
#pragma unroll
  for (int k = 0; k < 25; ++k) {
    int y = hc + k / 5 - 2, x = wc + k % 5 - 2;
    vk[k] = (y >= 0) && (y < 64) && (x >= 0) && (x < 64);
    int yc = y < 0 ? 0 : (y > 63 ? 63 : y);
    int xc = x < 0 ? 0 : (x > 63 ? 63 : x);
    offk[k] = (((b << 6) + yc) * 64 + xc) * 256;
  }
  bf16* orow = cu + (size_t)p * 256;
  for (int cb = 0; cb < 256; cb += 4) {
    v4f a = {};
#pragma unroll
    for (int k = 0; k < 25; ++k) {
      if (vk[k]) {
        v4f v = *(const v4f*)(coarse + offk[k] + cb);
        a += m[k] * v;
      }
    }
    orow[cb + 0] = (bf16)a.x;
    orow[cb + 1] = (bf16)a.y;
    orow[cb + 2] = (bf16)a.z;
    orow[cb + 3] = (bf16)a.w;
  }
}

// --------------------------- GEMM1: c_out = cu x W1 + b1 -------------------
// Block = 8 waves, all on the same N-half; that half's B pack (64 KB) staged
// in LDS. Each wave: 16 rows x 128 cols.
__global__ __launch_bounds__(256, 1)
void gemm_proj1_kernel(const bf16* __restrict__ A,   // [65536,256]
                       const bf16* __restrict__ Bp,
                       const float* __restrict__ bias,
                       float* __restrict__ Cf32, int Mtiles) {
  constexpr int KT = 8, NT = 8, N = 256;
  __shared__ bf16 bsh[NT * KT * 512];      // 64 KB
  int nh = blockIdx.x & 1;                 // N half: cols [nh*128, nh*128+128)
  stage_b(Bp + (size_t)nh * NT * KT * 512, bsh, NT * KT * 1024);

  int wid  = threadIdx.x >> 5;
  int lane = threadIdx.x & 31;
  int mtile = (blockIdx.x >> 1) * 8 + wid;
  if (mtile >= Mtiles) return;
  int row0 = mtile * 16;
  v8f acc[NT];
#pragma unroll
  for (int n = 0; n < NT; ++n) { v8f z = {}; acc[n] = z; }
  for (int kt = 0; kt < KT; ++kt) {
    v16bf a = load_a_bf16(A, N, row0, kt * 32, lane);
#pragma unroll
    for (int n = 0; n < NT; ++n) {
      v16bf b = load_b_lds(bsh, n * KT + kt, lane);
      acc[n] = __builtin_amdgcn_wmma_f32_16x16x32_bf16(
          false, a, false, b, (short)0, acc[n], false, false);
    }
  }
  int r = lane & 15, h = lane >> 4;
#pragma unroll
  for (int n = 0; n < NT; ++n) {
    int col = (nh * NT + n) * 16 + r;
    float bv = bias[col];
#pragma unroll
    for (int g = 0; g < 8; ++g) {
      int row = row0 + g + 8 * h;
      Cf32[(size_t)row * N + col] = acc[n][g] + bv;
    }
  }
}

// ------------- GEMM2: f_out = c_out x W2 + b2, fused gate blend ------------
__global__ __launch_bounds__(256, 1)
void gemm_proj2_kernel(const float* __restrict__ A,   // c_out f32
                       const bf16* __restrict__ Bp,
                       const float* __restrict__ bias,
                       const float* __restrict__ gate, // [4,64,64]
                       float* __restrict__ out, int Mtiles) {
  constexpr int KT = 8, NT = 8, N = 256;
  __shared__ bf16 bsh[NT * KT * 512];      // 64 KB
  int nh = blockIdx.x & 1;
  stage_b(Bp + (size_t)nh * NT * KT * 512, bsh, NT * KT * 1024);

  int wid  = threadIdx.x >> 5;
  int lane = threadIdx.x & 31;
  int mtile = (blockIdx.x >> 1) * 8 + wid;
  if (mtile >= Mtiles) return;
  int row0 = mtile * 16;
  v8f acc[NT];
#pragma unroll
  for (int n = 0; n < NT; ++n) { v8f z = {}; acc[n] = z; }
  for (int kt = 0; kt < KT; ++kt) {
    v16bf a = load_a_f32(A, N, row0, kt * 32, lane);  // f32->bf16 on the fly
#pragma unroll
    for (int n = 0; n < NT; ++n) {
      v16bf b = load_b_lds(bsh, n * KT + kt, lane);
      acc[n] = __builtin_amdgcn_wmma_f32_16x16x32_bf16(
          false, a, false, b, (short)0, acc[n], false, false);
    }
  }
  int r = lane & 15, h = lane >> 4;
  float bv[NT];
#pragma unroll
  for (int n = 0; n < NT; ++n) bv[n] = bias[(nh * NT + n) * 16 + r];
#pragma unroll
  for (int g = 0; g < 8; ++g) {
    int row = row0 + g + 8 * h;
    int b = row >> 14, rem = row & 16383, i = rem >> 7, j = rem & 127;
    float gt = gate[((b << 6) + (i >> 1)) * 64 + (j >> 1)];
#pragma unroll
    for (int n = 0; n < NT; ++n) {
      int col = (nh * NT + n) * 16 + r;
      size_t idx = (size_t)row * N + col;
      float fo = acc[n][g] + bv[n];
      out[idx] = gt * fo + (1.f - gt) * A[idx];
    }
  }
}

// ---------------------------------------------------------------------------
extern "C" void kernel_launch(void* const* d_in, const int* in_sizes, int n_in,
                              void* d_out, int out_size, void* d_ws, size_t ws_size,
                              hipStream_t stream) {
  (void)in_sizes; (void)n_in; (void)out_size; (void)ws_size;
  constexpr int Bn = 4, H = 64, W = 64, FH = 128, FW = 128;
  constexpr int Cf = 128, Cc = 256, E = 64, F = 256;
  const int Mf = Bn * FH * FW;   // 65536 fine pixels
  const int Mc = Bn * H * W;     // 16384 coarse pixels

  const float* fine          = (const float*)d_in[0];
  const float* coarse        = (const float*)d_in[1];
  const float* gate_w        = (const float*)d_in[2];
  const float* gate_b        = (const float*)d_in[3];
  const float* ss_fine_w     = (const float*)d_in[4];
  const float* ss_fine_b     = (const float*)d_in[5];
  const float* ss_coarse_w   = (const float*)d_in[6];
  const float* ss_content_w  = (const float*)d_in[7];
  const float* ss_content_b  = (const float*)d_in[8];
  const float* coarse_proj_w = (const float*)d_in[9];
  const float* coarse_proj_b = (const float*)d_in[10];
  const float* fine_proj_w   = (const float*)d_in[11];
  const float* fine_proj_b   = (const float*)d_in[12];
  float* out = (float*)d_out;

  // workspace carve (256 B aligned; every buffer is written before read)
  uint8_t* ws = (uint8_t*)d_ws;
  size_t off = 0;
  auto alloc = [&](size_t bytes) -> void* {
    void* p = ws + off;
    off += (bytes + 255) & ~(size_t)255;
    return p;
  };
  float* gate    = (float*)alloc((size_t)Mc * 4);            //  64 KB
  float* f_emb   = (float*)alloc((size_t)Mf * E * 4);        //  16 MB
  float* c_emb   = (float*)alloc((size_t)Mc * E * 4);        //   4 MB
  float* f_log   = (float*)alloc((size_t)Mf * 25 * 4);       // 6.4 MB
  float* c_log   = (float*)alloc((size_t)Mc * 25 * 4);       // 1.6 MB
  bf16*  cu_bf   = (bf16*) alloc((size_t)Mf * Cc * 2);       //  32 MB
  float* cout    = (float*)alloc((size_t)Mf * F * 4);        //  64 MB
  bf16*  pk_fine = (bf16*) alloc((size_t)Cf * E * 2);
  bf16*  pk_crs  = (bf16*) alloc((size_t)Cc * E * 2);
  bf16*  pk_w1   = (bf16*) alloc((size_t)Cc * F * 2);
  bf16*  pk_w2   = (bf16*) alloc((size_t)F * F * 2);

  // 1) pack all 1x1 weights into WMMA B-fragment layout
  pack_weights_kernel<<<(Cf * E + 255) / 256, 256, 0, stream>>>(ss_fine_w, pk_fine, Cf, E);
  pack_weights_kernel<<<(Cc * E + 255) / 256, 256, 0, stream>>>(ss_coarse_w, pk_crs, Cc, E);
  pack_weights_kernel<<<(Cc * F + 255) / 256, 256, 0, stream>>>(coarse_proj_w, pk_w1, Cc, F);
  pack_weights_kernel<<<(F * F + 255) / 256, 256, 0, stream>>>(fine_proj_w, pk_w2, F, F);

  // 2) gate = sigmoid(coarse . gate_w + b) at coarse resolution
  gate_kernel<<<(Mc + 255) / 256, 256, 0, stream>>>(coarse, gate_w, gate_b, gate, Mc, Cc);

  // 3) embeddings via WMMA (K=128 -> 4 ktiles / K=256 -> 8 ktiles; N=64 -> 4 ntiles)
  gemm_a32_kernel<4, 4, true ><<<Mf / 16 / 8, 256, 0, stream>>>(fine,   pk_fine, ss_fine_b, f_emb, Mf / 16, Cf, E);
  gemm_a32_kernel<8, 4, false><<<Mc / 16 / 8, 256, 0, stream>>>(coarse, pk_crs,  nullptr,   c_emb, Mc / 16, Cc, E);

  // 4) shared 3x3 content conv at both resolutions
  conv3x3_kernel<<<(Mf + 255) / 256, 256, 0, stream>>>(f_emb, ss_content_w, ss_content_b, f_log, Bn, FH, FW);
  conv3x3_kernel<<<(Mc + 255) / 256, 256, 0, stream>>>(c_emb, ss_content_w, ss_content_b, c_log, Bn, H, W);

  // 5) CARAFE reassembly with fused softmax (kernel = f + upsample2x(c))
  carafe_kernel<<<(Mf + 255) / 256, 256, 0, stream>>>(coarse, f_log, c_log, cu_bf);

  // 6) c_out = cu x W1 + b1   (WMMA bf16, f32 accumulate; N split in halves)
  gemm_proj1_kernel<<<Mf / 16 / 8 * 2, 256, 0, stream>>>(cu_bf, pk_w1, coarse_proj_b, cout, Mf / 16);

  // 7) out = gate*(c_out x W2 + b2) + (1-gate)*c_out  (fused epilogue)
  gemm_proj2_kernel<<<Mf / 16 / 8 * 2, 256, 0, stream>>>(cout, pk_w2, fine_proj_b, gate, out, Mf / 16);
}
